// SelfAttention_42545946034782
// MI455X (gfx1250) — compile-verified, matrix operand order checked by reading, not run on hardware
//
#include <hip/hip_runtime.h>
#include <stdint.h>

typedef __attribute__((ext_vector_type(16))) __bf16 v16bf;
typedef __attribute__((ext_vector_type(8)))  float  v8f;
typedef __attribute__((ext_vector_type(4)))  unsigned int u32x4;
typedef __attribute__((ext_vector_type(8)))  int i32x8;
typedef __attribute__((ext_vector_type(4)))  int i32x4;

#define B_  4
#define S_  2048
#define E_  1024
#define H_  16
#define D_  64

__device__ __forceinline__ __bf16 to_bf16(float x)  { return (__bf16)x; }
__device__ __forceinline__ __bf16 to_bf16(__bf16 x) { return x; }
__device__ __forceinline__ void store_out(float* p, float v)  { *p = v; }
__device__ __forceinline__ void store_out(__bf16* p, float v) { *p = (__bf16)v; }

// Opaque zero offset + memory clobber: keeps the __shared__ base pointer in
// addrspace(3) (so reads lower to ds_load, not flat_load) while preventing
// the optimizer from folding/reordering loads of TDM-written LDS.
__device__ __forceinline__ unsigned lds_fence_zero() {
  unsigned x = 0;
  asm volatile("" : "+v"(x) :: "memory");
  return x;
}

// 2D tile load via Tensor Data Mover: rows x (width_dw dwords), global row
// pitch stride_dw dwords, packed contiguously into LDS at lds_byte_addr.
// D# layout per CDNA5 ISA ch.8 (group0: count/lds/global/type, group1: dims).
__device__ __forceinline__ void tdm_load_2d(uint32_t lds_byte_addr,
                                            const void* gptr,
                                            uint32_t width_dw, uint32_t rows,
                                            uint32_t stride_dw) {
  const uint64_t ga = (uint64_t)(uintptr_t)gptr;
  u32x4 g0;
  g0[0] = 1u;                                              // count=1, user desc
  g0[1] = lds_byte_addr;                                   // lds_addr
  g0[2] = (uint32_t)ga;                                    // global_addr[31:0]
  g0[3] = (uint32_t)((ga >> 32) & 0x01ffffffu) | (2u << 30); // [56:32] | type=2
  i32x8 g1;
  g1[0] = (int)(2u << 16);                                 // data_size = 4B
  g1[1] = (int)((width_dw & 0xffffu) << 16);               // tensor_dim0 lo
  g1[2] = (int)((width_dw >> 16) | ((rows & 0xffffu) << 16)); // d0 hi | d1 lo
  g1[3] = (int)((rows >> 16) | ((width_dw & 0xffffu) << 16)); // d1 hi | tile_dim0
  g1[4] = (int)(rows & 0xffffu);                           // tile_dim1, tile_dim2=0
  g1[5] = (int)stride_dw;                                  // tensor_dim0_stride lo
  g1[6] = 0;
  g1[7] = 0;
  const i32x4 z4 = {0, 0, 0, 0};
#if defined(__clang_major__) && (__clang_major__ >= 23)
  const i32x8 z8 = {0, 0, 0, 0, 0, 0, 0, 0};
  __builtin_amdgcn_tensor_load_to_lds(g0, g1, z4, z4, z8, 0);
#else
  __builtin_amdgcn_tensor_load_to_lds(g0, g1, z4, z4, 0);
#endif
}

// ---------------------------------------------------------------------------
// f32 -> bf16 weight pre-conversion (runs once; weights reused 64x per GEMM)
__global__ __launch_bounds__(256) void cvt_bf16_kernel(
    const float* __restrict__ src, __bf16* __restrict__ dst, int n) {
  const int i = blockIdx.x * 256 + threadIdx.x;
  if (i < n) dst[i] = (__bf16)src[i];
}

// ---------------------------------------------------------------------------
// C[M,N] = A[M,K] @ Wbf[N,K]^T + bias[N]
// 8 waves/block; wave w -> rows [bx*128+w*16,+16), cols [by*64,+64).
// W tile (64 x 32 bf16) staged into LDS by TDM, double-buffered.
template <typename InT, typename OutT>
__global__ __launch_bounds__(256) void gemm_bias_kernel(
    const InT* __restrict__ A, const __bf16* __restrict__ Wbf,
    const float* __restrict__ bias, OutT* __restrict__ C,
    int M, int N, int K)
{
  __shared__ __bf16 wtile[2][64 * 32];

  const int lane   = threadIdx.x & 31;
  const int wave   = threadIdx.x >> 5;
  const int laneHi = lane >> 4;
  const int ln     = lane & 15;
  const int mBase  = blockIdx.x * 128 + wave * 16;
  const int nBase  = blockIdx.y * 64;
  const int niter  = K / 32;

  if (wave == 0)
    tdm_load_2d((uint32_t)(uintptr_t)&wtile[0][0],
                Wbf + (size_t)nBase * K, 16u, 64u, (uint32_t)(K / 2));

  v8f acc[4] = {v8f{}, v8f{}, v8f{}, v8f{}};
  const InT* arow = A + (size_t)(mBase + ln) * K + (laneHi ? 8 : 0);
  int cur = 0;

  for (int i = 0; i < niter; ++i) {
    const int kb = i * 32;
    if (wave == 0) {
      const int kn = (i + 1 < niter) ? kb + 32 : kb;   // dummy re-load on last
      tdm_load_2d((uint32_t)(uintptr_t)&wtile[cur ^ 1][0],
                  Wbf + (size_t)nBase * K + kn, 16u, 64u, (uint32_t)(K / 2));
      __builtin_amdgcn_s_wait_tensorcnt((short)1);     // current buffer landed
    }
    __syncthreads();                                   // publish wtile[cur]

    // ---- A fragment (16x32 bf16, ISA lane layout) ----
    v16bf a;
    {
      const InT* ap = arow + kb;
      __builtin_prefetch(ap + 32, 0, 0);
#pragma unroll
      for (int j = 0; j < 16; j += 2) {
        const int k = (j < 8) ? j : j + 8;
        a[j]     = to_bf16(ap[k]);
        a[j + 1] = to_bf16(ap[k + 1]);
      }
    }
    // ---- 4 B fragments from LDS via ds_load (bf16, contiguous along k) ----
    const unsigned fz = lds_fence_zero();
    const __bf16* wt = &wtile[cur][0];
#pragma unroll
    for (int f = 0; f < 4; ++f) {
      const __bf16* wp = wt + fz + (f * 16 + ln) * 32 + (laneHi ? 8 : 0);
      v16bf b;
#pragma unroll
      for (int j = 0; j < 16; j += 2) {
        const int k = (j < 8) ? j : j + 8;
        b[j]     = wp[k];
        b[j + 1] = wp[k + 1];
      }
      acc[f] = __builtin_amdgcn_wmma_f32_16x16x32_bf16(
          false, a, false, b, (short)0, acc[f], false, false);
    }
    __syncthreads();                                   // done with wtile[cur]
    cur ^= 1;
  }

  // ---- epilogue: bias + store (C/D layout: VGPR r -> row r / r+8) ----
#pragma unroll
  for (int f = 0; f < 4; ++f) {
    const int col = nBase + f * 16 + ln;
    const float bv = bias[col];
#pragma unroll
    for (int r = 0; r < 8; ++r) {
      const int row = mBase + r + (laneHi ? 8 : 0);
      store_out(C + (size_t)row * N + col, acc[f][r] + bv);
    }
  }
}

// ---------------------------------------------------------------------------
// Flash attention: one wave per (b, h, 16-query block). Per-wave K/V tiles
// (32 keys x 64 dims bf16) staged into private LDS slabs by TDM, double-buffered.
__global__ __launch_bounds__(256) void flash_attn_kernel(
    const __bf16* __restrict__ Qp, const __bf16* __restrict__ Kp,
    const __bf16* __restrict__ Vp, const int* __restrict__ mask,
    __bf16* __restrict__ O)
{
  __shared__ __bf16 ktile[8][2][32 * 64];
  __shared__ __bf16 vtile[8][2][32 * 64];
  __shared__ __bf16 pstage[8][16 * 32];

  const int lane   = threadIdx.x & 31;
  const int laneHi = lane >> 4;
  const int ln     = lane & 15;
  const int wv = __builtin_amdgcn_readfirstlane((int)(threadIdx.x >> 5));
  const int gw = __builtin_amdgcn_readfirstlane((int)(blockIdx.x * 8 + (threadIdx.x >> 5)));

  const int qblocks = S_ / 16;                 // 128
  const int b   = gw / (H_ * qblocks);
  const int rem = gw % (H_ * qblocks);
  const int h   = rem / qblocks;
  const int q0  = (rem % qblocks) * 16;

  const size_t kvbase = (size_t)b * S_ * E_ + h * D_;   // element offset, key 0

  // ---- prime first K/V tiles via TDM ----
  tdm_load_2d((uint32_t)(uintptr_t)&ktile[wv][0][0], Kp + kvbase, 32u, 32u, E_ / 2);
  tdm_load_2d((uint32_t)(uintptr_t)&vtile[wv][0][0], Vp + kvbase, 32u, 32u, E_ / 2);

  // ---- Q fragments (2 chunks of K=32 over head_dim 64), direct global ----
  v16bf qa[2];
  {
    const size_t qrow = ((size_t)b * S_ + q0 + ln) * E_ + h * D_;
#pragma unroll
    for (int c = 0; c < 2; ++c) {
      const __bf16* qp = Qp + qrow + c * 32 + (laneHi ? 8 : 0);
#pragma unroll
      for (int j = 0; j < 16; j += 2) {
        const int k = (j < 8) ? j : j + 8;
        qa[c][j]     = qp[k];
        qa[c][j + 1] = qp[k + 1];
      }
    }
  }

  v8f o[4] = {v8f{}, v8f{}, v8f{}, v8f{}};
  float mrow[8], lrow[8];
#pragma unroll
  for (int r = 0; r < 8; ++r) { mrow[r] = -3.0e38f; lrow[r] = 0.0f; }

  const float scale = 0.03125f;   // 1/sqrt(1024)
  __bf16* st = pstage[wv];
  int cur = 0;

  for (int k0 = 0; k0 < S_; k0 += 32) {
    // ---- issue next K/V tiles, then wait for current pair ----
    const int kn = (k0 + 32 < S_) ? k0 + 32 : k0;       // dummy on last iter
    tdm_load_2d((uint32_t)(uintptr_t)&ktile[wv][cur ^ 1][0],
                Kp + kvbase + (size_t)kn * E_, 32u, 32u, E_ / 2);
    tdm_load_2d((uint32_t)(uintptr_t)&vtile[wv][cur ^ 1][0],
                Vp + kvbase + (size_t)kn * E_, 32u, 32u, E_ / 2);
    __builtin_amdgcn_s_wait_tensorcnt((short)2);

    const unsigned fz = lds_fence_zero();
    const __bf16* kt = &ktile[wv][cur][0];
    const __bf16* vt = &vtile[wv][cur][0];

    // ---- scores: 16 queries x 32 keys, 4 WMMAs, K-frags from LDS ----
    v8f s0 = {}, s1 = {};
#pragma unroll
    for (int c = 0; c < 2; ++c) {
#pragma unroll
      for (int t = 0; t < 2; ++t) {
        const __bf16* kp = kt + fz + (t * 16 + ln) * 64 + c * 32 + (laneHi ? 8 : 0);
        v16bf kb;
#pragma unroll
        for (int j = 0; j < 16; j += 2) {
          const int k = (j < 8) ? j : j + 8;
          kb[j]     = kp[k];
          kb[j + 1] = kp[k + 1];
        }
        if (t == 0) s0 = __builtin_amdgcn_wmma_f32_16x16x32_bf16(
                        false, qa[c], false, kb, (short)0, s0, false, false);
        else        s1 = __builtin_amdgcn_wmma_f32_16x16x32_bf16(
                        false, qa[c], false, kb, (short)0, s1, false, false);
      }
    }

    // ---- mask + scale ----
    const int* mrow_p = mask + (size_t)b * S_ * S_;
#pragma unroll
    for (int r = 0; r < 8; ++r) {
      const int row = q0 + r + (laneHi ? 8 : 0);
      const int c0  = k0 + ln;
      const int mv0 = mrow_p[(size_t)row * S_ + c0];
      const int mv1 = mrow_p[(size_t)row * S_ + c0 + 16];
      s0[r] = (mv0 == 0) ? -3.0e18f : s0[r] * scale;
      s1[r] = (mv1 == 0) ? -3.0e18f : s1[r] * scale;
    }

    // ---- online softmax ----
    float alpha[8], mnew[8];
#pragma unroll
    for (int r = 0; r < 8; ++r) {
      float rm = fmaxf(s0[r], s1[r]);
#pragma unroll
      for (int off = 1; off < 16; off <<= 1)
        rm = fmaxf(rm, __shfl_xor(rm, off, 32));
      mnew[r]  = fmaxf(mrow[r], rm);
      alpha[r] = expf(mrow[r] - mnew[r]);
      mrow[r]  = mnew[r];
    }
#pragma unroll
    for (int r = 0; r < 8; ++r) {
      s0[r] = expf(s0[r] - mnew[r]);
      s1[r] = expf(s1[r] - mnew[r]);
      float rs = s0[r] + s1[r];
#pragma unroll
      for (int off = 1; off < 16; off <<= 1)
        rs += __shfl_xor(rs, off, 32);
      lrow[r] = lrow[r] * alpha[r] + rs;
    }
#pragma unroll
    for (int f = 0; f < 4; ++f)
#pragma unroll
      for (int r = 0; r < 8; ++r) o[f][r] *= alpha[r];

    // ---- stage P: C-layout -> LDS -> A-layout ----
#pragma unroll
    for (int r = 0; r < 8; ++r) {
      const int rl = r + (laneHi ? 8 : 0);
      st[rl * 32 + ln]      = (__bf16)s0[r];
      st[rl * 32 + 16 + ln] = (__bf16)s1[r];
    }
    asm volatile("s_wait_dscnt 0x0" ::: "memory");
    v16bf pa;
    {
      const __bf16* ps = st + ln * 32 + (laneHi ? 8 : 0);
#pragma unroll
      for (int j = 0; j < 16; j += 2) {
        const int k = (j < 8) ? j : j + 8;
        pa[j]     = ps[k];
        pa[j + 1] = ps[k + 1];
      }
    }

    // ---- O += P @ V : 4 WMMAs, V-frags from LDS ----
#pragma unroll
    for (int f = 0; f < 4; ++f) {
      v16bf vb;
#pragma unroll
      for (int j = 0; j < 16; ++j) {
        const int kk = ((j < 8) ? j : j + 8) + (laneHi ? 8 : 0);
        vb[j] = vt[fz + kk * 64 + f * 16 + ln];
      }
      o[f] = __builtin_amdgcn_wmma_f32_16x16x32_bf16(
          false, pa, false, vb, (short)0, o[f], false, false);
    }
    cur ^= 1;
  }

  // ---- finalize: O /= l ----
#pragma unroll
  for (int r = 0; r < 8; ++r) {
    const float inv = 1.0f / lrow[r];
    const int row = q0 + r + (laneHi ? 8 : 0);
    __bf16* op = O + ((size_t)b * S_ + row) * E_ + h * D_;
#pragma unroll
    for (int f = 0; f < 4; ++f)
      op[f * 16 + ln] = (__bf16)(o[f][r] * inv);
  }
}

// ---------------------------------------------------------------------------
extern "C" void kernel_launch(void* const* d_in, const int* in_sizes, int n_in,
                              void* d_out, int out_size, void* d_ws, size_t ws_size,
                              hipStream_t stream) {
  (void)in_sizes; (void)n_in; (void)out_size; (void)ws_size;

  const float* q    = (const float*)d_in[0];
  const float* k    = (const float*)d_in[1];
  const float* v    = (const float*)d_in[2];
  const int*   mask = (const int*)  d_in[3];
  const float* wq_w = (const float*)d_in[4];
  const float* wq_b = (const float*)d_in[5];
  const float* wk_w = (const float*)d_in[6];
  const float* wk_b = (const float*)d_in[7];
  const float* wv_w = (const float*)d_in[8];
  const float* wv_b = (const float*)d_in[9];
  const float* fo_w = (const float*)d_in[10];
  const float* fo_b = (const float*)d_in[11];
  float* out = (float*)d_out;

  const size_t TOT  = (size_t)B_ * S_ * E_;   // 8,388,608
  const size_t WTOT = (size_t)E_ * E_;        // 1,048,576
  __bf16* qp  = (__bf16*)d_ws;
  __bf16* kp  = qp  + TOT;
  __bf16* vp  = kp  + TOT;
  __bf16* at  = vp  + TOT;
  __bf16* wqb = at  + TOT;
  __bf16* wkb = wqb + WTOT;
  __bf16* wvb = wkb + WTOT;
  __bf16* wob = wvb + WTOT;

  const int cvtBlocks = (int)(WTOT / 256);
  cvt_bf16_kernel<<<cvtBlocks, 256, 0, stream>>>(wq_w, wqb, (int)WTOT);
  cvt_bf16_kernel<<<cvtBlocks, 256, 0, stream>>>(wk_w, wkb, (int)WTOT);
  cvt_bf16_kernel<<<cvtBlocks, 256, 0, stream>>>(wv_w, wvb, (int)WTOT);
  cvt_bf16_kernel<<<cvtBlocks, 256, 0, stream>>>(fo_w, wob, (int)WTOT);

  const int M = B_ * S_, N = E_, K = E_;
  dim3 ggrid(M / 128, N / 64);                // (64, 16)
  dim3 gblk(256);

  gemm_bias_kernel<float, __bf16><<<ggrid, gblk, 0, stream>>>(q, wqb, wq_b, qp, M, N, K);
  gemm_bias_kernel<float, __bf16><<<ggrid, gblk, 0, stream>>>(k, wkb, wk_b, kp, M, N, K);
  gemm_bias_kernel<float, __bf16><<<ggrid, gblk, 0, stream>>>(v, wvb, wv_b, vp, M, N, K);

  const int totalWaves = B_ * H_ * (S_ / 16); // 8192
  flash_attn_kernel<<<totalWaves / 8, 256, 0, stream>>>(qp, kp, vp, mask, at);

  gemm_bias_kernel<__bf16, float><<<ggrid, gblk, 0, stream>>>(at, wob, fo_b, out, M, N, K);
}
